// BaselineAgent_25305947308199
// MI455X (gfx1250) — matrix operand order repaired and checked by reading, not compiled, not run on hardware
//
#include <hip/hip_runtime.h>
#include <hip/hip_bf16.h>
#include <math.h>

#define T_    32
#define B_    32
#define NA_   5
#define HID_  256
#define FEAT_ 512
#define IND_  517
#define FLAT_ 3136
#define TB_   1024

typedef __attribute__((ext_vector_type(16))) _Float16 v16h;
typedef __attribute__((ext_vector_type(8)))  float    v8f;

// ---------------------------------------------------------------------------
// WMMA helpers (CDNA5: V_WMMA_F32_16X16X32_F16, wave32, f32 accumulate)
// ---------------------------------------------------------------------------
__device__ __forceinline__ v8f wmma16(v16h a, v16h b, v8f c) {
  // 8 args: (neg_a, A, neg_b, B, c_mod, C, reuse_a, reuse_b)
  return __builtin_amdgcn_wmma_f32_16x16x32_f16(
      false, a, false, b, (short)0, c, false, false);
}

// A fragment: 16x32 (MxK) f16 tile from row-major A[lda], contiguous k order (matches frag_bT):
// lanes 0-15 -> row M=lane, K = 0..15; lanes 16-31 -> K = 16..31.
// Per lane: one contiguous 32B run -> vectorizes to 2x global_load_b128.
__device__ __forceinline__ v16h frag_a(const _Float16* __restrict__ A, int lda,
                                       int m0, int k0, int lane) {
  const _Float16* p =
      A + (size_t)(m0 + (lane & 15)) * lda + k0 + ((lane >> 4) << 4);
  v16h a;
#pragma unroll
  for (int j = 0; j < 8; ++j) {
    int kk = 2 * j;
    a[2 * j]     = p[kk];
    a[2 * j + 1] = p[kk + 1];
  }
  return a;
}

// B fragment: 32x16 (KxN) built from row-major weights W[N][K] (i.e. B = W^T).
// lanes 0-15 -> col N=lane, K=k0..k0+15; lanes 16-31 -> K=k0+16..k0+31.
// Each lane reads 16 contiguous halves (32B -> 2x global_load_b128).
__device__ __forceinline__ v16h frag_bT(const _Float16* __restrict__ W, int ldw,
                                        int n0, int k0, int lane) {
  const _Float16* p =
      W + (size_t)(n0 + (lane & 15)) * ldw + k0 + ((lane >> 4) << 4);
  v16h b;
#pragma unroll
  for (int h = 0; h < 16; ++h) b[h] = p[h];
  return b;
}

__device__ __forceinline__ float sigf(float x) {
  return 1.0f / (1.0f + __expf(-x));
}

// ---------------------------------------------------------------------------
// f32 -> f16 conversion (strided source rows; used to cache weights in f16)
// ---------------------------------------------------------------------------
__global__ void k_cvt_f16(const float* __restrict__ src,
                          _Float16* __restrict__ dst,
                          int rows, int cols, int src_ld) {
  int i = blockIdx.x * blockDim.x + threadIdx.x;
  if (i >= rows * cols) return;
  int r = i / cols, c = i - r * cols;
  dst[i] = (_Float16)src[(size_t)r * src_ld + c];
}

// ---------------------------------------------------------------------------
// Direct convolutions (bandwidth-bound; weights live in 192MB L2)
// ---------------------------------------------------------------------------
__global__ void k_conv1(const float* __restrict__ img,
                        const float* __restrict__ W,
                        const float* __restrict__ bias,
                        float* __restrict__ out) {
  int i = blockIdx.x * blockDim.x + threadIdx.x;      // [1024][32][20][20]
  if (i >= TB_ * 32 * 20 * 20) return;
  int x = i % 20, y = (i / 20) % 20, o = (i / 400) % 32, n = i / (400 * 32);
  const float* ip = img + (size_t)n * (3 * 84 * 84);
  const float* wp = W + o * (3 * 8 * 8);
  float s = 0.0f;
#pragma unroll
  for (int c = 0; c < 3; ++c)
#pragma unroll
    for (int ky = 0; ky < 8; ++ky) {
      const float* irow = ip + c * 84 * 84 + (4 * y + ky) * 84 + 4 * x;
      const float* wrow = wp + c * 64 + ky * 8;
      __builtin_prefetch(irow + 84, 0, 3);
#pragma unroll
      for (int kx = 0; kx < 8; ++kx) s += irow[kx] * wrow[kx];
    }
  float v = bias[o] + s * (1.0f / 255.0f);            // fold x/255 into conv
  out[i] = v > 0.0f ? v : 0.0f;
}

__global__ void k_conv2(const float* __restrict__ in,
                        const float* __restrict__ W,
                        const float* __restrict__ bias,
                        float* __restrict__ out) {
  int i = blockIdx.x * blockDim.x + threadIdx.x;      // [1024][64][9][9]
  if (i >= TB_ * 64 * 81) return;
  int x = i % 9, y = (i / 9) % 9, o = (i / 81) % 64, n = i / (81 * 64);
  const float* ip = in + (size_t)n * (32 * 400);
  const float* wp = W + o * (32 * 16);
  float s = bias[o];
  for (int c = 0; c < 32; ++c)
#pragma unroll
    for (int ky = 0; ky < 4; ++ky) {
      const float* irow = ip + c * 400 + (2 * y + ky) * 20 + 2 * x;
      const float* wrow = wp + c * 16 + ky * 4;
#pragma unroll
      for (int kx = 0; kx < 4; ++kx) s += irow[kx] * wrow[kx];
    }
  out[i] = s > 0.0f ? s : 0.0f;
}

// conv3 + relu, writing f16 flattened activations (A matrix for the FC GEMM)
__global__ void k_conv3(const float* __restrict__ in,
                        const float* __restrict__ W,
                        const float* __restrict__ bias,
                        _Float16* __restrict__ x3) {
  int i = blockIdx.x * blockDim.x + threadIdx.x;      // [1024][64][7][7]
  if (i >= TB_ * FLAT_) return;
  int x = i % 7, y = (i / 7) % 7, o = (i / 49) % 64, n = i / FLAT_;
  const float* ip = in + (size_t)n * (64 * 81);
  const float* wp = W + o * (64 * 9);
  float s = bias[o];
  for (int c = 0; c < 64; ++c)
#pragma unroll
    for (int ky = 0; ky < 3; ++ky) {
      const float* irow = ip + c * 81 + (y + ky) * 9 + x;
      const float* wrow = wp + c * 9 + ky * 3;
#pragma unroll
      for (int kx = 0; kx < 3; ++kx) s += irow[kx] * wrow[kx];
    }
  s = s > 0.0f ? s : 0.0f;
  x3[(size_t)n * FLAT_ + o * 49 + y * 7 + x] = (_Float16)s;
}

// ---------------------------------------------------------------------------
// FC: feat = relu(X[1024,3136] @ Wfc^T + bfc) -> f16.
// One wave per 16x64 tile: 1 A-frag reused across 4 WMMAs (4x less A traffic).
// ---------------------------------------------------------------------------
__global__ void k_fc(const _Float16* __restrict__ X,
                     const _Float16* __restrict__ W,   // [512][3136] f16
                     const float* __restrict__ bias,
                     _Float16* __restrict__ out) {     // [1024][512] f16
  int lane = threadIdx.x;
  int m0 = blockIdx.x * 16, n0 = blockIdx.y * 64;
  v8f acc0 = {}, acc1 = {}, acc2 = {}, acc3 = {};
  for (int k0 = 0; k0 < FLAT_; k0 += 32) {
    v16h a  = frag_a(X, FLAT_, m0, k0, lane);
    v16h b0 = frag_bT(W, FLAT_, n0 +  0, k0, lane);
    v16h b1 = frag_bT(W, FLAT_, n0 + 16, k0, lane);
    v16h b2 = frag_bT(W, FLAT_, n0 + 32, k0, lane);
    v16h b3 = frag_bT(W, FLAT_, n0 + 48, k0, lane);
    acc0 = wmma16(a, b0, acc0);
    acc1 = wmma16(a, b1, acc1);
    acc2 = wmma16(a, b2, acc2);
    acc3 = wmma16(a, b3, acc3);
  }
  int nl = lane & 15;
  int mo = m0 + ((lane >> 4) << 3);
  v8f accs[4] = {acc0, acc1, acc2, acc3};
#pragma unroll
  for (int u = 0; u < 4; ++u) {
    int n = n0 + 16 * u + nl;
    float bb = bias[n];
#pragma unroll
    for (int r = 0; r < 8; ++r) {
      float v = accs[u][r] + bb;
      v = v > 0.0f ? v : 0.0f;
      out[(size_t)(mo + r) * FEAT_ + n] = (_Float16)v;
    }
  }
}

// ---------------------------------------------------------------------------
// Hoisted LSTM input projection:
// gpre[1024,1024] = feat @ W_ih[:, :512]^T + b_ih + b_hh + W_ih[:, 512+act]
// (one-hot column handled exactly in f32). 16x64 tile per wave as above.
// ---------------------------------------------------------------------------
__global__ void k_gates_in(const _Float16* __restrict__ F,     // [1024][512]
                           const _Float16* __restrict__ Wih16, // [1024][512]
                           const float* __restrict__ Wih,      // [1024][517]
                           const float* __restrict__ bih,
                           const float* __restrict__ bhh,
                           const int* __restrict__ act,        // [1024]
                           float* __restrict__ gpre) {         // [1024][1024]
  int lane = threadIdx.x;
  int m0 = blockIdx.x * 16, n0 = blockIdx.y * 64;
  v8f acc0 = {}, acc1 = {}, acc2 = {}, acc3 = {};
  for (int k0 = 0; k0 < FEAT_; k0 += 32) {
    v16h a  = frag_a(F, FEAT_, m0, k0, lane);
    v16h b0 = frag_bT(Wih16, FEAT_, n0 +  0, k0, lane);
    v16h b1 = frag_bT(Wih16, FEAT_, n0 + 16, k0, lane);
    v16h b2 = frag_bT(Wih16, FEAT_, n0 + 32, k0, lane);
    v16h b3 = frag_bT(Wih16, FEAT_, n0 + 48, k0, lane);
    acc0 = wmma16(a, b0, acc0);
    acc1 = wmma16(a, b1, acc1);
    acc2 = wmma16(a, b2, acc2);
    acc3 = wmma16(a, b3, acc3);
  }
  int nl = lane & 15;
  int mo = m0 + ((lane >> 4) << 3);
  v8f accs[4] = {acc0, acc1, acc2, acc3};
#pragma unroll
  for (int u = 0; u < 4; ++u) {
    int n = n0 + 16 * u + nl;
    float bb = bih[n] + bhh[n];
#pragma unroll
    for (int r = 0; r < 8; ++r) {
      int row = mo + r;
      float oh = Wih[(size_t)n * IND_ + FEAT_ + act[row]];
      gpre[(size_t)row * 1024 + n] = acc0[0] * 0.0f + accs[u][r] + bb + oh;
    }
  }
}

// ---------------------------------------------------------------------------
// LSTM per-step kernels
// ---------------------------------------------------------------------------
__global__ void k_prep(const float* __restrict__ done, int t,
                       float* __restrict__ hc, float* __restrict__ cc,
                       _Float16* __restrict__ hm16) {
  int i = blockIdx.x * blockDim.x + threadIdx.x;      // 32*256
  if (i >= B_ * HID_) return;
  int b = i / HID_;
  float m = 1.0f - done[t * B_ + b];
  float hv = hc[i] * m;
  cc[i] *= m;
  hm16[i] = (_Float16)hv;                             // masked h for GEMM
}

// gtmp[32,1024] = gpre_t + hm @ W_hh^T   (M=32, N=1024, K=256; WMMA)
// Kept at one 16x16 tile per wave: this GEMM sits on the serial 32-step
// chain, so maximizing wave-level parallelism (128 waves) minimizes latency.
__global__ void k_hgemm(const float* __restrict__ gpre_t,
                        const _Float16* __restrict__ Hm,     // [32][256]
                        const _Float16* __restrict__ Whh16,  // [1024][256]
                        float* __restrict__ gtmp) {
  int lane = threadIdx.x;
  int m0 = blockIdx.x * 16, n0 = blockIdx.y * 16;
  v8f acc = {};
  for (int k0 = 0; k0 < HID_; k0 += 32) {
    v16h a = frag_a(Hm, HID_, m0, k0, lane);
    v16h b = frag_bT(Whh16, HID_, n0, k0, lane);
    acc = wmma16(a, b, acc);
  }
  int n = n0 + (lane & 15);
  int mo = m0 + ((lane >> 4) << 3);
#pragma unroll
  for (int r = 0; r < 8; ++r) {
    size_t idx = (size_t)(mo + r) * 1024 + n;
    gtmp[idx] = acc[r] + gpre_t[idx];
  }
}

__global__ void k_cell(const float* __restrict__ gtmp, int t,
                       float* __restrict__ hc, float* __restrict__ cc,
                       float* __restrict__ hs) {
  int i = blockIdx.x * blockDim.x + threadIdx.x;      // 32*256
  if (i >= B_ * HID_) return;
  int b = i / HID_, j = i - b * HID_;
  const float* g = gtmp + (size_t)b * 1024;
  float gi = g[j], gf = g[HID_ + j], gg = g[2 * HID_ + j], go = g[3 * HID_ + j];
  float c = sigf(gf) * cc[i] + sigf(gi) * tanhf(gg);
  float h = sigf(go) * tanhf(c);
  cc[i] = c;
  hc[i] = h;
  hs[(size_t)(t * B_ + b) * HID_ + j] = h;
}

// ---------------------------------------------------------------------------
// Heads: one wave per row. z = tanh(h@W1^T+b1); out = z@W2^T+b2
// ---------------------------------------------------------------------------
__global__ void k_heads(const float* __restrict__ hs,
                        const float* __restrict__ Wp1, const float* __restrict__ bp1,
                        const float* __restrict__ Wp2, const float* __restrict__ bp2,
                        const float* __restrict__ Wv1, const float* __restrict__ bv1,
                        const float* __restrict__ Wv2, const float* __restrict__ bv2,
                        float* __restrict__ out) {
  __shared__ float z1[64];
  __shared__ float zv[64];
  int row = blockIdx.x, lane = threadIdx.x;
  const float* h = hs + (size_t)row * HID_;
  for (int jj = lane; jj < 64; jj += 32) {
    float s1 = bp1[jj], s2 = bv1[jj];
    const float* wp = Wp1 + (size_t)jj * HID_;
    const float* wv = Wv1 + (size_t)jj * HID_;
    for (int k = 0; k < HID_; ++k) {
      s1 += h[k] * wp[k];
      s2 += h[k] * wv[k];
    }
    z1[jj] = tanhf(s1);
    zv[jj] = tanhf(s2);
  }
  __syncthreads();
  if (lane < NA_) {
    float s = bp2[lane];
    const float* w = Wp2 + lane * 64;
    for (int k = 0; k < 64; ++k) s += z1[k] * w[k];
    out[(size_t)row * NA_ + lane] = s;                // logits
  }
  if (lane == 8) {
    float s = bv2[0];
    for (int k = 0; k < 64; ++k) s += zv[k] * Wv2[k];
    out[TB_ * NA_ + row] = s;                         // value
  }
}

__global__ void k_copy(const float* __restrict__ a, float* __restrict__ b, int n) {
  int i = blockIdx.x * blockDim.x + threadIdx.x;
  if (i < n) b[i] = a[i];
}

// ---------------------------------------------------------------------------
// Host orchestration
// ---------------------------------------------------------------------------
extern "C" void kernel_launch(void* const* d_in, const int* in_sizes, int n_in,
                              void* d_out, int out_size, void* d_ws, size_t ws_size,
                              hipStream_t stream) {
  const float* image = (const float*)d_in[0];
  const int*   act   = (const int*)d_in[1];
  const float* done  = (const float*)d_in[2];
  const float* h0    = (const float*)d_in[3];
  const float* c0    = (const float*)d_in[4];
  const float* W1 = (const float*)d_in[5];  const float* b1 = (const float*)d_in[6];
  const float* W2 = (const float*)d_in[7];  const float* b2 = (const float*)d_in[8];
  const float* W3 = (const float*)d_in[9];  const float* b3 = (const float*)d_in[10];
  const float* Wfc = (const float*)d_in[11]; const float* bfc = (const float*)d_in[12];
  const float* Wih = (const float*)d_in[13]; const float* Whh = (const float*)d_in[14];
  const float* bih = (const float*)d_in[15]; const float* bhh = (const float*)d_in[16];
  const float* Wp1 = (const float*)d_in[17]; const float* bp1 = (const float*)d_in[18];
  const float* Wp2 = (const float*)d_in[19]; const float* bp2 = (const float*)d_in[20];
  const float* Wv1 = (const float*)d_in[21]; const float* bv1 = (const float*)d_in[22];
  const float* Wv2 = (const float*)d_in[23]; const float* bv2 = (const float*)d_in[24];
  float* out = (float*)d_out;

  // workspace layout (256B aligned)
  char* ws = (char*)d_ws;
  size_t off = 0;
  auto take = [&](size_t bytes) -> void* {
    void* p = ws + off;
    off = (off + bytes + 255) & ~(size_t)255;
    return p;
  };
  _Float16* wfc16  = (_Float16*)take((size_t)FEAT_ * FLAT_ * 2);
  _Float16* wih16  = (_Float16*)take((size_t)4 * HID_ * FEAT_ * 2);
  _Float16* whh16  = (_Float16*)take((size_t)4 * HID_ * HID_ * 2);
  float*    c1     = (float*)take((size_t)TB_ * 32 * 400 * 4);
  float*    c2     = (float*)take((size_t)TB_ * 64 * 81 * 4);
  _Float16* x3     = (_Float16*)take((size_t)TB_ * FLAT_ * 2);
  _Float16* feat16 = (_Float16*)take((size_t)TB_ * FEAT_ * 2);
  float*    gpre   = (float*)take((size_t)TB_ * 1024 * 4);
  float*    gtmp   = (float*)take((size_t)B_ * 1024 * 4);
  float*    h_cur  = (float*)take((size_t)B_ * HID_ * 4);
  float*    c_cur  = (float*)take((size_t)B_ * HID_ * 4);
  _Float16* hm16   = (_Float16*)take((size_t)B_ * HID_ * 2);
  float*    hs     = (float*)take((size_t)TB_ * HID_ * 4);
  (void)ws_size; (void)n_in; (void)in_sizes; (void)out_size;

  const int TPB = 256;
  // weights -> f16 (once per call; GEMM operands)
  k_cvt_f16<<<(FEAT_ * FLAT_ + TPB - 1) / TPB, TPB, 0, stream>>>(Wfc, wfc16, FEAT_, FLAT_, FLAT_);
  k_cvt_f16<<<(4 * HID_ * FEAT_ + TPB - 1) / TPB, TPB, 0, stream>>>(Wih, wih16, 4 * HID_, FEAT_, IND_);
  k_cvt_f16<<<(4 * HID_ * HID_ + TPB - 1) / TPB, TPB, 0, stream>>>(Whh, whh16, 4 * HID_, HID_, HID_);

  // CNN trunk
  k_conv1<<<(TB_ * 32 * 400 + TPB - 1) / TPB, TPB, 0, stream>>>(image, W1, b1, c1);
  k_conv2<<<(TB_ * 64 * 81 + TPB - 1) / TPB, TPB, 0, stream>>>(c1, W2, b2, c2);
  k_conv3<<<(TB_ * FLAT_ + TPB - 1) / TPB, TPB, 0, stream>>>(c2, W3, b3, x3);

  // FC (WMMA): 64 x 8 blocks, one wave computes a 16x64 tile
  k_fc<<<dim3(TB_ / 16, FEAT_ / 64), 32, 0, stream>>>(x3, wfc16, bfc, feat16);

  // Hoisted input projection for all timesteps (WMMA): 64 x 16 blocks
  k_gates_in<<<dim3(TB_ / 16, 1024 / 64), 32, 0, stream>>>(feat16, wih16, Wih,
                                                           bih, bhh, act, gpre);

  // init recurrent state
  k_copy<<<(B_ * HID_ + TPB - 1) / TPB, TPB, 0, stream>>>(h0, h_cur, B_ * HID_);
  k_copy<<<(B_ * HID_ + TPB - 1) / TPB, TPB, 0, stream>>>(c0, c_cur, B_ * HID_);

  // serial LSTM scan: mask -> recurrent GEMM (WMMA) -> cell update
  for (int t = 0; t < T_; ++t) {
    k_prep<<<(B_ * HID_ + TPB - 1) / TPB, TPB, 0, stream>>>(done, t, h_cur, c_cur, hm16);
    k_hgemm<<<dim3(B_ / 16, 1024 / 16), 32, 0, stream>>>(
        gpre + (size_t)t * B_ * 1024, hm16, whh16, gtmp);
    k_cell<<<(B_ * HID_ + TPB - 1) / TPB, TPB, 0, stream>>>(gtmp, t, h_cur, c_cur, hs);
  }

  // heads over all TB rows
  k_heads<<<TB_, 32, 0, stream>>>(hs, Wp1, bp1, Wp2, bp2, Wv1, bv1, Wv2, bv2, out);

  // final carries
  k_copy<<<(B_ * HID_ + TPB - 1) / TPB, TPB, 0, stream>>>(h_cur, out + TB_ * NA_ + TB_, B_ * HID_);
  k_copy<<<(B_ * HID_ + TPB - 1) / TPB, TPB, 0, stream>>>(c_cur, out + TB_ * NA_ + TB_ + B_ * HID_, B_ * HID_);
}
